// AdaPT_Conv2d_Systolic_50414326121233
// MI455X (gfx1250) — 8-bit matrix operand order repaired and checked by reading, compiled, not run on hardware
//
#include <hip/hip_runtime.h>

typedef int  v8i __attribute__((ext_vector_type(8)));
typedef int  v4i __attribute__((ext_vector_type(4)));

#define HH 224
#define WW 224
#define HWSZ (HH * WW)
#define CIN 32
#define COUT 64
#define NB 32
#define TPR 14
#define KBLKS 5

__global__ void init_amax(unsigned* amax) {
    if (threadIdx.x < 2) amax[threadIdx.x] = 0u;
}

__global__ void amax_reduce4(const float4* __restrict__ p, long long n4,
                             unsigned* __restrict__ out) {
    __shared__ unsigned sm[256];
    float m = 0.0f;
    for (long long i = (long long)blockIdx.x * blockDim.x + threadIdx.x; i < n4;
         i += (long long)gridDim.x * blockDim.x) {
        float4 v = p[i];
        m = fmaxf(m, fmaxf(fmaxf(fabsf(v.x), fabsf(v.y)),
                           fmaxf(fabsf(v.z), fabsf(v.w))));
    }
    sm[threadIdx.x] = __float_as_uint(m);
    __syncthreads();
    for (int s = 128; s > 0; s >>= 1) {
        if ((int)threadIdx.x < s) {
            unsigned a = sm[threadIdx.x], b = sm[threadIdx.x + s];
            sm[threadIdx.x] = a > b ? a : b;
        }
        __syncthreads();
    }
    if (threadIdx.x == 0) atomicMax(out, sm[0]);
}

__global__ void quant_pack_w(const float* __restrict__ w,
                             unsigned* __restrict__ wpack,
                             const unsigned* __restrict__ amax) {
    int idx = blockIdx.x * blockDim.x + threadIdx.x;
    if (idx >= 4 * KBLKS * 32 * 8) return;
    float s = 127.0f / __uint_as_float(amax[1]);
    int d      = idx & 7;
    int lane   = (idx >> 3) & 31;
    int k_blk  = (idx >> 8) % KBLKS;
    int m_tile = (idx >> 8) / KBLKS;
    int M = lane & 15, khalf = lane >> 4;
    unsigned pack = 0u;
    for (int b = 0; b < 4; ++b) {
        int K_local = ((d >> 2) * 32) + khalf * 16 +
                      ((d & 3) * 4) + b;
        int K = k_blk * 64 + K_local;
        int q = 0;
        if (K < 288) {
            int tap = K >> 5, cin = K & 31;
            int m = m_tile * 16 + M;
            float v = w[(m * CIN + cin) * 9 + tap];
            q = (int)fminf(127.0f, fmaxf(-127.0f, rintf(v * s)));
        }
        pack |= ((unsigned)(unsigned char)(signed char)q) << (8 * b);
    }
    wpack[idx] = pack;
}

__global__ void quant_x_nhwc(const float* __restrict__ x,
                             signed char* __restrict__ qx,
                             const unsigned* __restrict__ amax) {
    int t = blockIdx.x * blockDim.x + threadIdx.x;
    if (t >= NB * HWSZ) return;
    float s = 127.0f / __uint_as_float(amax[0]);
    int n = t / HWSZ, hw = t % HWSZ;
    const float* xp = x + (long long)n * CIN * HWSZ + hw;
    v4i d0, d1;
    for (int cg = 0; cg < 8; ++cg) {
        unsigned pack = 0u;
        for (int b = 0; b < 4; ++b) {
            int cin = cg * 4 + b;
            float v = xp[(long long)cin * HWSZ];
            float q = fminf(127.0f, fmaxf(-127.0f, rintf(v * s)));
            pack |= ((unsigned)(unsigned char)(signed char)(int)q) << (8 * b);
        }
        if (cg < 4) d0[cg] = (int)pack; else d1[cg - 4] = (int)pack;
    }
    v4i* op = (v4i*)(qx + (long long)t * 32);
    op[0] = d0;
    op[1] = d1;
}

__device__ __forceinline__ v4i load_tap(const signed char* __restrict__ qxn,
                                        int h, int iw_base, int tap, int khalf) {
    int kh = tap / 3, kw = tap % 3;
    int ih = h + kh - 1;
    int iw = iw_base + kw;
    v4i z = {0, 0, 0, 0};
    if ((unsigned)ih < (unsigned)HH && (unsigned)iw < (unsigned)WW)
        return *(const v4i*)(qxn + ((long long)ih * WW + iw) * 32 + khalf * 16);
    return z;
}

__global__ void conv_wmma_iu8(const signed char* __restrict__ qx,
                              const unsigned* __restrict__ wpack,
                              const float* __restrict__ bias,
                              const unsigned* __restrict__ amax,
                              float* __restrict__ out) {
    int wave = threadIdx.x >> 5;
    int lane = threadIdx.x & 31;
    int tile = blockIdx.x * 8 + wave;
    int n   = tile / (HH * TPR);
    int rem = tile % (HH * TPR);
    int h   = rem / TPR;
    int w0  = (rem % TPR) * 16;
    int col = lane & 15, khalf = lane >> 4;
    int iw_base = w0 + col - 1;
    const signed char* qxn = qx + (long long)n * HWSZ * 32;

    v8i acc[4];
#pragma unroll
    for (int mt = 0; mt < 4; ++mt) {
        v8i z = {0, 0, 0, 0, 0, 0, 0, 0};
        acc[mt] = z;
    }

#pragma unroll
    for (int kb = 0; kb < KBLKS; ++kb) {
        v4i b0 = load_tap(qxn, h, iw_base, kb * 2, khalf);
        v4i b1;
        if (kb * 2 + 1 < 9) b1 = load_tap(qxn, h, iw_base, kb * 2 + 1, khalf);
        else { v4i z4 = {0, 0, 0, 0}; b1 = z4; }
        v8i b = __builtin_shufflevector(b0, b1, 0, 1, 2, 3, 4, 5, 6, 7);
#pragma unroll
        for (int mt = 0; mt < 4; ++mt) {
            v8i a = *(const v8i*)(wpack + (((mt * KBLKS + kb) * 32 + lane) * 8));
            acc[mt] = __builtin_amdgcn_wmma_i32_16x16x64_iu8(true, a, true, b,
                                                             acc[mt],
                                                             false, false);
        }
    }

    float inv = (__uint_as_float(amax[0]) * (1.0f / 127.0f)) *
                (__uint_as_float(amax[1]) * (1.0f / 127.0f));
    long long obase = (long long)n * COUT * HWSZ + (long long)h * WW + (w0 + col);
#pragma unroll
    for (int mt = 0; mt < 4; ++mt) {
        int m0 = mt * 16 + khalf * 8;
#pragma unroll
        for (int r = 0; r < 8; ++r) {
            int m = m0 + r;
            float v = (float)acc[mt][r] * inv + bias[m];
            __builtin_nontemporal_store(v, &out[obase + (long long)m * HWSZ]);
        }
    }
}

extern "C" void kernel_launch(void* const* d_in, const int* in_sizes, int n_in,
                              void* d_out, int out_size, void* d_ws, size_t ws_size,
                              hipStream_t stream) {
    const float* x    = (const float*)d_in[0];
    const float* w    = (const float*)d_in[1];
    const float* bias = (const float*)d_in[2];
    float* out = (float*)d_out;

    unsigned*    amax  = (unsigned*)d_ws;
    unsigned*    wpack = (unsigned*)((char*)d_ws + 64);
    signed char* qx    = (signed char*)((char*)d_ws + 32768);

    const long long n_x4 = (long long)NB * CIN * HWSZ / 4;
    const long long n_w4 = (long long)COUT * CIN * 9 / 4;

    init_amax<<<1, 32, 0, stream>>>(amax);
    amax_reduce4<<<2048, 256, 0, stream>>>((const float4*)x, n_x4, amax + 0);
    amax_reduce4<<<18, 256, 0, stream>>>((const float4*)w, n_w4, amax + 1);
    quant_pack_w<<<(4 * KBLKS * 32 * 8 + 255) / 256, 256, 0, stream>>>(w, wpack, amax);
    quant_x_nhwc<<<(NB * HWSZ) / 256, 256, 0, stream>>>(x, qx, amax);
    conv_wmma_iu8<<<(NB * HH * TPR) / 8, 256, 0, stream>>>(qx, wpack, bias,
                                                           amax, out);
}
